// MultiSimilarityLoss_21388937134457
// MI455X (gfx1250) — matrix operand order repaired and checked by reading, not compiled, not run on hardware
//
#include <hip/hip_runtime.h>

#define BDIM 512          // feature dim D
#define NROWS 8192        // batch B
#define TM 128            // rows per workgroup (8 waves x 16)
#define COLS_PER_SLICE 1024
#define CT_PER_SLICE 64   // 16-wide col tiles per slice
#define LDSB_STRIDE 520   // 512 + 8 bf16 pad -> conflict-free ds_load_b128
#define NTILE (NROWS / 16)   // 512 tiles per dimension

static constexpr float kThresh = 0.5f;
static constexpr float kMargin = 0.1f;
static constexpr float kEps    = 1e-5f;

typedef __attribute__((ext_vector_type(4)))  unsigned int u32x4;
typedef __attribute__((ext_vector_type(8)))  float        v8f;
typedef __attribute__((ext_vector_type(16))) __bf16       bf16x16;

union Frag { bf16x16 v; u32x4 q[2]; };

__device__ __forceinline__ unsigned short f2bf(float x) {
  unsigned u = __float_as_uint(x);
  u += 0x7FFFu + ((u >> 16) & 1u);          // round-to-nearest-even
  return (unsigned short)(u >> 16);
}
// order-preserving float <-> uint encoding for integer atomic min/max
__device__ __forceinline__ unsigned encOrd(float x) {
  unsigned u = __float_as_uint(x);
  return (u & 0x80000000u) ? ~u : (u | 0x80000000u);
}
__device__ __forceinline__ float decOrd(unsigned e) {
  unsigned u = (e & 0x80000000u) ? (e ^ 0x80000000u) : ~e;
  return __uint_as_float(u);
}

// ---------------- kernel 1: bf16 convert + sum of squared norms + init ----
__global__ void msloss_prep(const float* __restrict__ feats,
                            unsigned short* __restrict__ fbf,
                            float* __restrict__ endnorm,
                            unsigned* __restrict__ mpe,
                            unsigned* __restrict__ mne,
                            float* __restrict__ psum,
                            float* __restrict__ nsum) {
  const int gid = blockIdx.x * blockDim.x + threadIdx.x;   // 1,048,576 threads
  const float4 f = reinterpret_cast<const float4*>(feats)[gid];
  uint2 packed;
  packed.x = (unsigned)f2bf(f.x) | ((unsigned)f2bf(f.y) << 16);
  packed.y = (unsigned)f2bf(f.z) | ((unsigned)f2bf(f.w) << 16);
  reinterpret_cast<uint2*>(fbf)[gid] = packed;

  float ss = f.x * f.x + f.y * f.y + f.z * f.z + f.w * f.w;
  #pragma unroll
  for (int m = 1; m < 32; m <<= 1) ss += __shfl_xor(ss, m);
  __shared__ float red[8];
  if ((threadIdx.x & 31) == 0) red[threadIdx.x >> 5] = ss;
  __syncthreads();
  if (threadIdx.x == 0) {
    float s = 0.f;
    #pragma unroll
    for (int i = 0; i < 8; i++) s += red[i];
    atomicAdd(endnorm, s);
  }
  if (gid < NROWS) {
    mpe[gid] = 0xFF800000u;   // encOrd(+inf): min identity
    mne[gid] = 0x007FFFFFu;   // encOrd(-inf): max identity
    psum[gid] = 0.f;
    nsum[gid] = 0.f;
  }
}

// ---------------- WMMA Gram pass ------------------------------------------
// PASS 0: per-row min over positives / max over negatives (atomic min/max);
//         if STORE, also writes each 16x16 sim tile (bf16, register-native
//         tile-blocked layout: 512 B/tile, lane stores its 8 D-values packed)
// PASS 1: per-row masked exp sums using pass-0 thresholds (atomic add)
template <int PASS, bool STORE>
__global__ __launch_bounds__(256, 1)
void msloss_sim_pass(const unsigned short* __restrict__ fbf,
                     const int* __restrict__ labels,
                     const float* __restrict__ endnorm,
                     unsigned* __restrict__ mpe,
                     unsigned* __restrict__ mne,
                     float* __restrict__ psum,
                     float* __restrict__ nsum,
                     unsigned short* __restrict__ simOut) {
  __shared__ unsigned short ldsB[2][16 * LDSB_STRIDE];  // double-buffered B
  __shared__ int ldsLab[COLS_PER_SLICE];                // 4 KB col labels

  const int tid  = threadIdx.x;
  const int wave = tid >> 5;
  const int lane = tid & 31;
  const int l16  = lane & 15;
  const int hi   = lane >> 4;

  const int rowbase = blockIdx.x * TM + wave * 16;
  const int colbase = blockIdx.y * COLS_PER_SLICE;

  const float invn = 1.0f / endnorm[0];

  for (int i = tid; i < COLS_PER_SLICE; i += 256) ldsLab[i] = labels[colbase + i];

  int labr[8];
  #pragma unroll
  for (int r = 0; r < 8; r++) labr[r] = labels[rowbase + hi * 8 + r];

  float minp[8], maxn[8];
  if (PASS == 1) {
    #pragma unroll
    for (int r = 0; r < 8; r++) {
      const int row = rowbase + hi * 8 + r;
      minp[r] = decOrd(mpe[row]);
      maxn[r] = decOrd(mne[row]);
    }
  }

  // Resident A fragments: 16 k-steps * 32B/lane = 128 VGPRs, reused for all
  // 64 column tiles of this slice.  A layout (16-bit A 16x32): lane holds row
  // M=lane%16, K chunks {b..b+7, b+16..b+23}, b = 8*(lane>=16).
  Frag a[16];
  {
    const unsigned short* arow = fbf + (size_t)(rowbase + l16) * BDIM;
    const int abase = hi ? 8 : 0;
    #pragma unroll
    for (int t = 0; t < 16; t++) {
      const int k0 = t * 32 + abase;
      a[t].q[0] = *reinterpret_cast<const u32x4*>(arow + k0);
      a[t].q[1] = *reinterpret_cast<const u32x4*>(arow + k0 + 16);
    }
  }

  float accA[8], accB[8];
  #pragma unroll
  for (int r = 0; r < 8; r++) {
    accA[r] = (PASS == 0) ?  __builtin_inff() : 0.f;
    accB[r] = (PASS == 0) ? -__builtin_inff() : 0.f;
  }

  // Per-thread cooperative staging geometry: 4 chunks of 8 bf16 per thread.
  int gOff[4], lOff[4];
  #pragma unroll
  for (int it = 0; it < 4; it++) {
    const int c = tid + it * 256;              // chunk id 0..1023
    const int row = c >> 6, cir = c & 63;
    gOff[it] = row * BDIM + cir * 8;           // within a 16xBDIM tile
    lOff[it] = row * LDSB_STRIDE + cir * 8;    // within one LDS buffer
  }

  u32x4 stg[4];                                // staging regs (global -> LDS)
  // prologue: tile 0 -> buffer 0
  {
    const unsigned short* src = fbf + (size_t)colbase * BDIM;
    #pragma unroll
    for (int it = 0; it < 4; it++)
      stg[it] = *reinterpret_cast<const u32x4*>(src + gOff[it]);
    #pragma unroll
    for (int it = 0; it < 4; it++)
      *reinterpret_cast<u32x4*>(&ldsB[0][lOff[it]]) = stg[it];
  }
  __syncthreads();

  // B layout (16-bit B 32x16): lane holds col N=lane%16, K chunks
  // {b..b+7, b+16..b+23}, b = 8*(lane>=16).  B[k][n] = feats[n][k] -> row-major read.
  const int bbase = hi ? 8 : 0;

  for (int ct = 0; ct < CT_PER_SLICE; ct++) {
    // issue global loads for tile ct+1 NOW; latency hidden by 16 WMMAs below
    if (ct + 1 < CT_PER_SLICE) {
      const unsigned short* src = fbf + (size_t)(colbase + (ct + 1) * 16) * BDIM;
      #pragma unroll
      for (int it = 0; it < 4; it++)
        stg[it] = *reinterpret_cast<const u32x4*>(src + gOff[it]);
    }

    const unsigned short* brow = &ldsB[ct & 1][l16 * LDSB_STRIDE + bbase];

    // 2-deep software pipeline of B-fragment LDS loads around the WMMA chain
    Frag bp[2];
    bp[0].q[0] = *reinterpret_cast<const u32x4*>(brow + 0 * 32);
    bp[0].q[1] = *reinterpret_cast<const u32x4*>(brow + 0 * 32 + 16);
    bp[1].q[0] = *reinterpret_cast<const u32x4*>(brow + 1 * 32);
    bp[1].q[1] = *reinterpret_cast<const u32x4*>(brow + 1 * 32 + 16);

    v8f d = {};
    #pragma unroll
    for (int t = 0; t < 16; t++) {
      d = __builtin_amdgcn_wmma_f32_16x16x32_bf16(false, a[t].v, false,
                                                  bp[t & 1].v, (short)0, d,
                                                  false, false);
      if (t + 2 < 16) {
        bp[t & 1].q[0] = *reinterpret_cast<const u32x4*>(brow + (t + 2) * 32);
        bp[t & 1].q[1] = *reinterpret_cast<const u32x4*>(brow + (t + 2) * 32 + 16);
      }
    }

    float simv[8];
    #pragma unroll
    for (int r = 0; r < 8; r++) simv[r] = d[r] * invn;

    const int labc = ldsLab[ct * 16 + l16];
    // D layout: VGPR r of lane -> sim[row rowbase+8*hi+r][col coltile+l16]
    #pragma unroll
    for (int r = 0; r < 8; r++) {
      const float sim = simv[r];
      const bool same = (labr[r] == labc);
      if (PASS == 0) {
        if (same && sim < 1.0f - kEps) accA[r] = fminf(accA[r], sim);
        if (!same)                     accB[r] = fmaxf(accB[r], sim);
      } else {
        if (same && (sim < 1.0f - kEps) && (sim - kMargin < maxn[r]))
          accA[r] += __expf(-2.0f * (sim - kThresh));
        if (!same && (sim + kMargin > minp[r]))
          accB[r] += __expf( 2.0f * (sim - kThresh));
      }
    }

    if (STORE) {
      // pack 8 sims as bf16 (truncation; values ~1e-4, margins 0.1 -> safe)
      u32x4 pk;
      #pragma unroll
      for (int i = 0; i < 4; i++)
        pk[i] = (__float_as_uint(simv[2 * i]) >> 16) |
                (__float_as_uint(simv[2 * i + 1]) & 0xFFFF0000u);
      const size_t tileIdx = (size_t)(rowbase >> 4) * NTILE + ((colbase >> 4) + ct);
      *reinterpret_cast<u32x4*>(simOut + tileIdx * 256 + lane * 8) = pk;
    }

    // store prefetched tile ct+1 into the other buffer, then one barrier
    if (ct + 1 < CT_PER_SLICE) {
      #pragma unroll
      for (int it = 0; it < 4; it++)
        *reinterpret_cast<u32x4*>(&ldsB[(ct + 1) & 1][lOff[it]]) = stg[it];
    }
    __syncthreads();
  }

  // reduce over columns: xor masks 1..8 stay within each 16-lane half
  #pragma unroll
  for (int r = 0; r < 8; r++) {
    float va = accA[r], vb = accB[r];
    #pragma unroll
    for (int m = 1; m < 16; m <<= 1) {
      const float oa = __shfl_xor(va, m);
      const float ob = __shfl_xor(vb, m);
      if (PASS == 0) { va = fminf(va, oa); vb = fmaxf(vb, ob); }
      else           { va += oa;           vb += ob; }
    }
    if (l16 == 0) {
      const int row = rowbase + hi * 8 + r;
      if (PASS == 0) {
        atomicMin(&mpe[row], encOrd(va));
        atomicMax(&mne[row], encOrd(vb));
      } else {
        atomicAdd(&psum[row], va);
        atomicAdd(&nsum[row], vb);
      }
    }
  }
}

// ---------------- streaming pass 2 over stored sim tiles ------------------
// grid (64 row-tile groups, 4 col slices of 2048); wave w owns row tile
// rt = bx*8+w and streams 128 consecutive tiles: pure b128 loads + VALU.
__global__ __launch_bounds__(256)
void msloss_pass2_stream(const unsigned short* __restrict__ simT,
                         const int* __restrict__ labels,
                         const unsigned* __restrict__ mpe,
                         const unsigned* __restrict__ mne,
                         float* __restrict__ psum,
                         float* __restrict__ nsum) {
  __shared__ int ldsLab[2048];
  const int tid = threadIdx.x, wave = tid >> 5, lane = tid & 31;
  const int l16 = lane & 15, hi = lane >> 4;
  const int rt = blockIdx.x * 8 + wave;          // row tile 0..511
  const int ctg0 = blockIdx.y * 128;             // first col tile of slice

  for (int i = tid; i < 2048; i += 256) ldsLab[i] = labels[blockIdx.y * 2048 + i];
  __syncthreads();

  const int rowbase = rt * 16;
  int labr[8]; float minp[8], maxn[8];
  #pragma unroll
  for (int r = 0; r < 8; r++) {
    const int row = rowbase + hi * 8 + r;
    labr[r] = labels[row];
    minp[r] = decOrd(mpe[row]);
    maxn[r] = decOrd(mne[row]);
  }

  float accA[8], accB[8];
  #pragma unroll
  for (int r = 0; r < 8; r++) { accA[r] = 0.f; accB[r] = 0.f; }

  const unsigned short* base = simT + ((size_t)rt * NTILE + ctg0) * 256 + lane * 8;
  for (int ct = 0; ct < 128; ct++) {
    const u32x4 q = *reinterpret_cast<const u32x4*>(base + ct * 256);
    const int labc = ldsLab[ct * 16 + l16];
    #pragma unroll
    for (int r = 0; r < 8; r++) {
      const unsigned u = q[r >> 1];
      const float sim = __uint_as_float((r & 1) ? (u & 0xFFFF0000u) : (u << 16));
      const bool same = (labr[r] == labc);
      if (same && (sim < 1.0f - kEps) && (sim - kMargin < maxn[r]))
        accA[r] += __expf(-2.0f * (sim - kThresh));
      if (!same && (sim + kMargin > minp[r]))
        accB[r] += __expf( 2.0f * (sim - kThresh));
    }
  }

  #pragma unroll
  for (int r = 0; r < 8; r++) {
    float va = accA[r], vb = accB[r];
    #pragma unroll
    for (int m = 1; m < 16; m <<= 1) { va += __shfl_xor(va, m); vb += __shfl_xor(vb, m); }
    if (l16 == 0) {
      const int row = rowbase + hi * 8 + r;
      atomicAdd(&psum[row], va);
      atomicAdd(&nsum[row], vb);
    }
  }
}

// ---------------- kernel 4: per-row loss + global sum ---------------------
__global__ void msloss_finalize(const float* __restrict__ psum,
                                const float* __restrict__ nsum,
                                float* __restrict__ out) {
  const int r = blockIdx.x * blockDim.x + threadIdx.x;
  const float ps = psum[r], ns = nsum[r];
  float v = 0.f;
  if (ps > 0.f && ns > 0.f)                 // valid row: both masks nonempty
    v = 0.5f * log1pf(ps) + 0.5f * log1pf(ns);
  #pragma unroll
  for (int m = 1; m < 32; m <<= 1) v += __shfl_xor(v, m);
  __shared__ float red[8];
  if ((threadIdx.x & 31) == 0) red[threadIdx.x >> 5] = v;
  __syncthreads();
  if (threadIdx.x == 0) {
    float s = 0.f;
    #pragma unroll
    for (int i = 0; i < 8; i++) s += red[i];
    atomicAdd(out, s * (1.0f / (float)NROWS));
  }
}

extern "C" void kernel_launch(void* const* d_in, const int* in_sizes, int n_in,
                              void* d_out, int out_size, void* d_ws, size_t ws_size,
                              hipStream_t stream) {
  const float* feats  = (const float*)d_in[0];
  const int*   labels = (const int*)d_in[1];
  float* out = (float*)d_out;
  char*  ws  = (char*)d_ws;

  // workspace layout
  unsigned short* fbf = (unsigned short*)ws;                       // 8 MB bf16 F
  float*    endnorm = (float*)(ws + (size_t)NROWS * BDIM * 2);
  unsigned* mpe  = (unsigned*)((char*)endnorm + 16);               // enc min_pos
  unsigned* mne  = mpe + NROWS;                                    // enc max_neg
  float*    psum = (float*)(mne + NROWS);
  float*    nsum = psum + NROWS;
  const size_t simOff   = (size_t)16 << 20;                        // 16 MB
  const size_t simBytes = (size_t)NROWS * NROWS * 2;               // 134 MB bf16
  unsigned short* simT  = (unsigned short*)(ws + simOff);
  const bool storeSim = ws_size >= simOff + simBytes;  // constant per harness

  hipMemsetAsync(out, 0, sizeof(float), stream);
  hipMemsetAsync(endnorm, 0, sizeof(float), stream);

  msloss_prep<<<(NROWS * BDIM / 4) / 256, 256, 0, stream>>>(
      feats, fbf, endnorm, mpe, mne, psum, nsum);

  dim3 grid(NROWS / TM, NROWS / COLS_PER_SLICE);   // 64 x 8 = 512 workgroups
  if (storeSim) {
    // pass 1 computes Gram once (WMMA) and spills bf16 sim tiles to L2-resident
    // workspace; pass 2 is a pure streaming re-read (no matrix recompute).
    msloss_sim_pass<0, true><<<grid, 256, 0, stream>>>(
        fbf, labels, endnorm, mpe, mne, psum, nsum, simT);
    msloss_pass2_stream<<<dim3(64, 4), 256, 0, stream>>>(
        simT, labels, mpe, mne, psum, nsum);
  } else {
    msloss_sim_pass<0, false><<<grid, 256, 0, stream>>>(
        fbf, labels, endnorm, mpe, mne, psum, nsum, simT);
    msloss_sim_pass<1, false><<<grid, 256, 0, stream>>>(
        fbf, labels, endnorm, mpe, mne, psum, nsum, simT);
  }

  msloss_finalize<<<NROWS / 256, 256, 0, stream>>>(psum, nsum, out);
}
